// Net_678604833378
// MI455X (gfx1250) — matrix operand order repaired and checked by reading, not compiled, not run on hardware
//
#include <hip/hip_runtime.h>
#include <hip/hip_bf16.h>
#include <stdint.h>

// ---------------- problem constants ----------------
#define BB   32
#define NN   512
#define DD   768
#define EE   4096
#define MM   (BB * NN)          // 16384 rows for the GEMM
#define ETOT (EE + NN)          // edges + self loops

typedef __attribute__((ext_vector_type(8)))  __bf16 v8bf;
typedef __attribute__((ext_vector_type(16))) __bf16 v16bf;
typedef __attribute__((ext_vector_type(8)))  float  v8f;

__device__ __forceinline__ v16bf join16(v8bf lo, v8bf hi) {
  return __builtin_shufflevector(lo, hi, 0,1,2,3,4,5,6,7,8,9,10,11,12,13,14,15);
}

// ---------------- small prep kernels ----------------
__global__ void k_deg(const int* __restrict__ dst, const float* __restrict__ ew,
                      float* __restrict__ deg) {
  int e = blockIdx.x * 256 + threadIdx.x;
  if (e >= ETOT) return;
  int d; float w;
  if (e < EE) { d = dst[e]; w = ew[e]; } else { d = e - EE; w = 1.0f; }
  atomicAdd(&deg[d], w);
}

__global__ void k_dinv(const float* __restrict__ deg, float* __restrict__ dinv) {
  int n = blockIdx.x * 256 + threadIdx.x;
  if (n >= NN) return;
  float v = deg[n];
  dinv[n] = (v > 0.0f) ? rsqrtf(v) : 0.0f;
}

__global__ void k_norm(const int* __restrict__ src, const int* __restrict__ dst,
                       const float* __restrict__ ew, const float* __restrict__ dinv,
                       float* __restrict__ nrm) {
  int e = blockIdx.x * 256 + threadIdx.x;
  if (e >= ETOT) return;
  int s, d; float w;
  if (e < EE) { s = src[e]; d = dst[e]; w = ew[e]; }
  else        { s = d = e - EE;          w = 1.0f; }
  nrm[e] = dinv[s] * w * dinv[d];
}

__global__ void k_f32_to_bf16(const float* __restrict__ in, __bf16* __restrict__ out, int n) {
  int i = blockIdx.x * 256 + threadIdx.x;
  if (i < n) out[i] = (__bf16)in[i];
}

// ---------------- WMMA GEMM: C[M,N] = A[M,K] * B[K,N] (bf16 in, f32 acc) ----
// Block tile: 128 (M) x 128 (N), K-step 32. 256 threads = 8 waves (wave32).
// Wave (mw, nw) with mw = wave&3, nw = wave>>2 owns a 32x64 register tile:
// 2 M-subtiles x 4 N-subtiles = 8 WMMAs per K-step (A frags reused over 4 N
// tiles, B frags reused over 2 M tiles -> 1.5 ds_load_b128 per WMMA).
#define TM  128
#define TN  128
#define TK  32
#define LDA 40   // bf16 elements per LDS row (80B, 16B aligned)
#define LDB 40

__global__ __launch_bounds__(256)
void k_gemm_bf16(const __bf16* __restrict__ A, const __bf16* __restrict__ Bw,
                 float* __restrict__ C, int M, int Nn, int K) {
  __shared__ __align__(16) __bf16 As[TM * LDA];      // As[m][k]
  __shared__ __align__(16) __bf16 Bs[TN * LDB];      // Bs[n][k] (transposed W tile)

  const int tid  = threadIdx.x;
  const int wave = tid >> 5;
  const int lane = tid & 31;
  const int half = lane >> 4;      // K-half selector (ISA 16-bit operand layout)
  const int ln   = lane & 15;      // row (A) / column (B,C) index
  const int mw   = wave & 3;       // M subtile group (32 rows)
  const int nw   = wave >> 2;      // N half (64 cols)

  const int bm = blockIdx.y * TM;
  const int bn = blockIdx.x * TN;

  v8f acc00 = {}, acc01 = {}, acc02 = {}, acc03 = {};
  v8f acc10 = {}, acc11 = {}, acc12 = {}, acc13 = {};

  // A staging: thread -> row tid/2, 16 contiguous bf16 at col (tid&1)*16
  const int ar = tid >> 1;
  const int ac = (tid & 1) * 16;
  // B staging: thread -> k-row tid/8 (0..31), 16 contiguous bf16 at col (tid&7)*16
  const int kr = tid >> 3;
  const int kc = (tid & 7) * 16;

  for (int kk = 0; kk < K; kk += TK) {
    // ---- stage A tile (row-major) ----
    const v8bf* pa = (const v8bf*)(A + (size_t)(bm + ar) * K + kk + ac);
    *(v8bf*)(As + ar * LDA + ac)     = pa[0];
    *(v8bf*)(As + ar * LDA + ac + 8) = pa[1];
    // ---- stage B tile transposed: Bs[n][k] ----
    const v8bf* pb = (const v8bf*)(Bw + (size_t)(kk + kr) * Nn + bn + kc);
    v8bf bv0 = pb[0];
    v8bf bv1 = pb[1];
#pragma unroll
    for (int j = 0; j < 8; ++j) {
      Bs[(kc + j)     * LDB + kr] = bv0[j];
      Bs[(kc + 8 + j) * LDB + kr] = bv1[j];
    }
    __syncthreads();

    // ---- build fragments per ISA 16-bit layouts ----
    // A 16x32: lanes 0-15 hold K 0-7 & 16-23, lanes 16-31 hold K 8-15 & 24-31
    const int am0 = mw * 32 + ln;
    const int am1 = mw * 32 + 16 + ln;
    v16bf a0 = join16(*(const v8bf*)(As + am0 * LDA + half * 8),
                      *(const v8bf*)(As + am0 * LDA + 16 + half * 8));
    v16bf a1 = join16(*(const v8bf*)(As + am1 * LDA + half * 8),
                      *(const v8bf*)(As + am1 * LDA + 16 + half * 8));
    // B 32x16: lane column n=ln, lanes 0-15 K 0-7 & 16-23, lanes 16-31 K 8-15 & 24-31
    const int nb = nw * 64 + ln;
    v16bf b0 = join16(*(const v8bf*)(Bs + (nb +  0) * LDB + half * 8),
                      *(const v8bf*)(Bs + (nb +  0) * LDB + 16 + half * 8));
    v16bf b1 = join16(*(const v8bf*)(Bs + (nb + 16) * LDB + half * 8),
                      *(const v8bf*)(Bs + (nb + 16) * LDB + 16 + half * 8));
    v16bf b2 = join16(*(const v8bf*)(Bs + (nb + 32) * LDB + half * 8),
                      *(const v8bf*)(Bs + (nb + 32) * LDB + 16 + half * 8));
    v16bf b3 = join16(*(const v8bf*)(Bs + (nb + 48) * LDB + half * 8),
                      *(const v8bf*)(Bs + (nb + 48) * LDB + 16 + half * 8));

    acc00 = __builtin_amdgcn_wmma_f32_16x16x32_bf16(false, a0, false, b0, (short)0, acc00, false, false);
    acc01 = __builtin_amdgcn_wmma_f32_16x16x32_bf16(false, a0, false, b1, (short)0, acc01, false, false);
    acc02 = __builtin_amdgcn_wmma_f32_16x16x32_bf16(false, a0, false, b2, (short)0, acc02, false, false);
    acc03 = __builtin_amdgcn_wmma_f32_16x16x32_bf16(false, a0, false, b3, (short)0, acc03, false, false);
    acc10 = __builtin_amdgcn_wmma_f32_16x16x32_bf16(false, a1, false, b0, (short)0, acc10, false, false);
    acc11 = __builtin_amdgcn_wmma_f32_16x16x32_bf16(false, a1, false, b1, (short)0, acc11, false, false);
    acc12 = __builtin_amdgcn_wmma_f32_16x16x32_bf16(false, a1, false, b2, (short)0, acc12, false, false);
    acc13 = __builtin_amdgcn_wmma_f32_16x16x32_bf16(false, a1, false, b3, (short)0, acc13, false, false);
    __syncthreads();
  }

  // D layout: VGPR i -> row half*8 + i, column ln
  const int rb0 = bm + mw * 32;          // first M subtile
  const int rb1 = bm + mw * 32 + 16;     // second M subtile
  const int cb  = bn + nw * 64 + ln;
#pragma unroll
  for (int i = 0; i < 8; ++i) {
    float* c0 = C + (size_t)(rb0 + half * 8 + i) * Nn + cb;
    c0[0]  = acc00[i];
    c0[16] = acc01[i];
    c0[32] = acc02[i];
    c0[48] = acc03[i];
    float* c1 = C + (size_t)(rb1 + half * 8 + i) * Nn + cb;
    c1[0]  = acc10[i];
    c1[16] = acc11[i];
    c1[32] = acc12[i];
    c1[48] = acc13[i];
  }
}

// ---------------- edge scatter-add: agg[b,dst,:] += nrm[e] * h[b,src,:] -----
__global__ __launch_bounds__(256)
void k_scatter(const float* __restrict__ h, const float* __restrict__ nrm,
               const int* __restrict__ src, const int* __restrict__ dst,
               float* __restrict__ agg) {
  const int e = blockIdx.x;
  const int b = blockIdx.y;
  int s, d;
  if (e < EE) { s = src[e]; d = dst[e]; } else { s = d = e - EE; }
  const float w = nrm[e];
  const float* __restrict__ hrow = h   + ((size_t)b * NN + s) * DD;
  float* __restrict__       arow = agg + ((size_t)b * NN + d) * DD;
  for (int c = threadIdx.x; c < DD; c += 256)
    atomicAdd(&arow[c], w * hrow[c]);
}

// ---------------- bias + ReLU epilogues ----------------
__global__ void k_bias_relu_bf16(const float* __restrict__ agg, const float* __restrict__ bias,
                                 __bf16* __restrict__ out) {
  int i = blockIdx.x * 256 + threadIdx.x;
  if (i >= MM * DD) return;
  float v = agg[i] + bias[i % DD];
  out[i] = (__bf16)fmaxf(v, 0.0f);
}

__global__ void k_bias_relu_f32(const float* __restrict__ agg, const float* __restrict__ bias,
                                float* __restrict__ out) {
  int i = blockIdx.x * 256 + threadIdx.x;
  if (i >= MM * DD) return;
  float v = agg[i] + bias[i % DD];
  out[i] = fmaxf(v, 0.0f);
}

// ---------------- host driver ----------------
extern "C" void kernel_launch(void* const* d_in, const int* in_sizes, int n_in,
                              void* d_out, int out_size, void* d_ws, size_t ws_size,
                              hipStream_t stream) {
  const float* x   = (const float*)d_in[0];
  const int*   ei  = (const int*)  d_in[1];   // [2, E] int32 (JAX x64 off)
  const float* ew  = (const float*)d_in[2];
  const float* W1  = (const float*)d_in[3];
  const float* b1  = (const float*)d_in[4];
  const float* W2  = (const float*)d_in[5];
  const float* b2  = (const float*)d_in[6];
  float* out = (float*)d_out;

  const int* src = ei;
  const int* dst = ei + EE;

  // -------- workspace carve-out (256B aligned) --------
  size_t off = 0;
  auto carve = [&](size_t bytes) -> char* {
    char* p = (char*)d_ws + off;
    off = (off + bytes + 255) & ~(size_t)255;
    return p;
  };
  float*  deg  = (float*) carve(NN * sizeof(float));
  float*  dinv = (float*) carve(NN * sizeof(float));
  float*  nrm  = (float*) carve(ETOT * sizeof(float));
  __bf16* W1b  = (__bf16*)carve((size_t)DD * DD * sizeof(__bf16));
  __bf16* W2b  = (__bf16*)carve((size_t)DD * DD * sizeof(__bf16));
  __bf16* ab   = (__bf16*)carve((size_t)MM * DD * sizeof(__bf16));  // activations bf16
  float*  g    = (float*) carve((size_t)MM * DD * sizeof(float));   // GEMM output
  float*  agg  = (float*) carve((size_t)MM * DD * sizeof(float));   // scatter accumulator

  const int threads = 256;
  const int nMD     = MM * DD;                 // 12,582,912
  const int gMD     = (nMD + threads - 1) / threads;
  const int nW      = DD * DD;
  const int gW      = (nW + threads - 1) / threads;
  const dim3 gemmGrid(DD / TN, MM / TM);       // (6, 128)
  const dim3 scatGrid(ETOT, BB);               // (4608, 32)

  // -------- degree / normalization (shared by both layers) --------
  hipMemsetAsync(deg, 0, NN * sizeof(float), stream);
  k_deg <<<(ETOT + threads - 1) / threads, threads, 0, stream>>>(dst, ew, deg);
  k_dinv<<<(NN + threads - 1) / threads, threads, 0, stream>>>(deg, dinv);
  k_norm<<<(ETOT + threads - 1) / threads, threads, 0, stream>>>(src, dst, ew, dinv, nrm);

  // -------- bf16 conversions --------
  k_f32_to_bf16<<<gMD, threads, 0, stream>>>(x,  ab,  nMD);
  k_f32_to_bf16<<<gW,  threads, 0, stream>>>(W1, W1b, nW);
  k_f32_to_bf16<<<gW,  threads, 0, stream>>>(W2, W2b, nW);

  // -------- layer 1 --------
  k_gemm_bf16<<<gemmGrid, threads, 0, stream>>>(ab, W1b, g, MM, DD, DD);
  hipMemsetAsync(agg, 0, (size_t)nMD * sizeof(float), stream);
  k_scatter<<<scatGrid, threads, 0, stream>>>(g, nrm, src, dst, agg);
  k_bias_relu_bf16<<<gMD, threads, 0, stream>>>(agg, b1, ab);   // ab now holds relu(h1) in bf16

  // -------- layer 2 --------
  k_gemm_bf16<<<gemmGrid, threads, 0, stream>>>(ab, W2b, g, MM, DD, DD);
  hipMemsetAsync(agg, 0, (size_t)nMD * sizeof(float), stream);
  k_scatter<<<scatGrid, threads, 0, stream>>>(g, nrm, src, dst, agg);
  k_bias_relu_f32<<<gMD, threads, 0, stream>>>(agg, b2, out);
}
